// Infomax_91173565760013
// MI455X (gfx1250) — compile-verified, matrix operand order checked by reading, not run on hardware
//
#include <hip/hip_runtime.h>
#include <hip/hip_bf16.h>

typedef __attribute__((ext_vector_type(16))) _Float16 v16h;
typedef __attribute__((ext_vector_type(8)))  float    v8f;

#define HD 128   // feature/hidden dim
#define TM 16    // rows per block tile
#define APAD 8   // f16 LDS pad (half elems)
#define OPAD 4   // f32 LDS pad

// ---------------------------------------------------------------- utilities
__global__ void zero_kernel(float* __restrict__ p, long long n) {
  long long i = (long long)blockIdx.x * blockDim.x + threadIdx.x;
  long long stride = (long long)gridDim.x * blockDim.x;
  for (; i < n; i += stride) p[i] = 0.0f;
}

__global__ void count_kernel(const int* __restrict__ dst, float* __restrict__ cnt, int E) {
  int e = blockIdx.x * blockDim.x + threadIdx.x;
  if (e < E) atomicAdd(&cnt[dst[e]], 1.0f);
}

// one wave (32 lanes) per edge: gather float4, scatter-add into L2-resident agg
__global__ __launch_bounds__(256)
void scatter_kernel(const float* __restrict__ xin,
                    const int* __restrict__ src,
                    const int* __restrict__ dst,
                    const int* __restrict__ perm,
                    float* __restrict__ agg, int E) {
  int gt = blockIdx.x * blockDim.x + threadIdx.x;
  int e = gt >> 5;
  int lane = gt & 31;
  if (e >= E) return;
  int s = src[e];
  int d = dst[e];
  int row = perm ? perm[s] : s;
  const float4 v = *(const float4*)(xin + (long long)row * HD + lane * 4);
  float* ap = agg + (long long)d * HD + lane * 4;
  atomicAdd(ap + 0, v.x);
  atomicAdd(ap + 1, v.y);
  atomicAdd(ap + 2, v.z);
  atomicAdd(ap + 3, v.w);
}

// Pre-pack a 128x128 f32 weight (row-major [k][n]) into per-lane WMMA B-fragment
// order (f16): frag (cb,kb4) -> 32 lanes x 16 halves contiguous.
// lane: n = cb*16 + (lane&15), hi = lane>>4; element i -> k = kb4*32 + hi*8 + (i&7) + (i>=8?16:0)
__global__ void pack_weight_kernel(const float* __restrict__ W, _Float16* __restrict__ out) {
  for (int it = threadIdx.x; it < 1024; it += blockDim.x) {
    int lane = it & 31;
    int kb4  = (it >> 5) & 3;
    int cb   = it >> 7;
    int hi   = lane >> 4;
    int n    = cb * 16 + (lane & 15);
    int k0   = kb4 * 32 + hi * 8;
    _Float16* dp = out + (((cb * 4 + kb4) * 32 + lane) * 16);
#pragma unroll
    for (int i = 0; i < 8; ++i) dp[i]     = (_Float16)W[(k0 + i) * HD + n];
#pragma unroll
    for (int i = 0; i < 8; ++i) dp[8 + i] = (_Float16)W[(k0 + 16 + i) * HD + n];
  }
}

// ------------------------------------------- fused SAGE layer (WMMA f16->f32)
// out[r,:] = prelu( l2norm( mean[r,:]@Wl + bias + xin[r,:]@Wr ), p )
__global__ __launch_bounds__(256)
void sage_wmma_kernel(const float* __restrict__ xin, const int* __restrict__ perm,
                      const float* __restrict__ agg, const float* __restrict__ cnt,
                      const _Float16* __restrict__ Wlp, const _Float16* __restrict__ Wrp,
                      const float* __restrict__ bl, const float* __restrict__ prelu_a,
                      float* __restrict__ out, float* __restrict__ colsum, int N) {
  __shared__ _Float16 sMean[TM * (HD + APAD)];
  __shared__ _Float16 sX[TM * (HD + APAD)];
  __shared__ float    sOut[TM * (HD + OPAD)];
  __shared__ float    sInv[TM];

  int tid = threadIdx.x;
  int r0 = blockIdx.x * TM;

  // stage mean and self tiles into LDS as f16
#pragma unroll
  for (int i = 0; i < 8; ++i) {
    int idx = tid + 256 * i;            // 0..2047
    int row = idx >> 7;
    int col = idx & (HD - 1);
    int g = r0 + row; if (g >= N) g = N - 1;
    float c = cnt[g];
    float m = agg[(long long)g * HD + col] / fmaxf(c, 1.0f);
    sMean[row * (HD + APAD) + col] = (_Float16)m;
    int srow = perm ? perm[g] : g;
    sX[row * (HD + APAD) + col] = (_Float16)xin[(long long)srow * HD + col];
  }
  __syncthreads();

  int wid = tid >> 5;          // wave -> 16-column block
  int lane = tid & 31;
  int hi = lane >> 4;
  int i16 = lane & 15;
  int colBase = wid * 16;

  v8f acc = {};
#pragma unroll
  for (int kb4 = 0; kb4 < 4; ++kb4) {
    int k0 = kb4 * 32 + hi * 8;
    const _Float16* mrow = sMean + i16 * (HD + APAD);
    const _Float16* xrow = sX    + i16 * (HD + APAD);
    v16h am, ax;
#pragma unroll
    for (int i = 0; i < 8; ++i) { am[i] = mrow[k0 + i]; am[8 + i] = mrow[k0 + 16 + i]; }
#pragma unroll
    for (int i = 0; i < 8; ++i) { ax[i] = xrow[k0 + i]; ax[8 + i] = xrow[k0 + 16 + i]; }
    const v16h bL = *(const v16h*)(Wlp + (((wid * 4 + kb4) * 32 + lane) * 16));
    const v16h bR = *(const v16h*)(Wrp + (((wid * 4 + kb4) * 32 + lane) * 16));
    acc = __builtin_amdgcn_wmma_f32_16x16x32_f16(false, am, false, bL, (short)0, acc, false, false);
    acc = __builtin_amdgcn_wmma_f32_16x16x32_f16(false, ax, false, bR, (short)0, acc, false, false);
  }

  // D layout: lane,vgpr r -> (M = r + 8*hi, Ncol = colBase + i16)
  float bias = bl[colBase + i16];
#pragma unroll
  for (int r = 0; r < 8; ++r) {
    int row = r + hi * 8;
    sOut[row * (HD + OPAD) + colBase + i16] = acc[r] + bias;
  }
  __syncthreads();

  if (tid < TM) {
    const float* orow = sOut + tid * (HD + OPAD);
    float s = 0.0f;
#pragma unroll 4
    for (int c = 0; c < HD; ++c) { float v = orow[c]; s += v * v; }
    sInv[tid] = 1.0f / fmaxf(sqrtf(s), 1e-12f);
  }
  __syncthreads();

  if (tid < HD) {
    int col = tid;
    float a = prelu_a[col];
    float csum = 0.0f;
    for (int row = 0; row < TM; ++row) {
      float v = sOut[row * (HD + OPAD) + col] * sInv[row];
      v = (v >= 0.0f) ? v : a * v;
      if (r0 + row < N) {
        out[(long long)(r0 + row) * HD + col] = v;
        csum += v;
      }
    }
    if (colsum) atomicAdd(&colsum[col], csum);
  }
}

// ---------------------------------------------------------- summary & logits
__global__ void summary_kernel(const float* __restrict__ colsum, const float* __restrict__ Wd,
                               float* __restrict__ wvec, int N) {
  __shared__ float s_sh[HD];
  int tid = threadIdx.x;
  float m = colsum[tid] / (float)N;
  s_sh[tid] = 1.0f / (1.0f + expf(-m));
  __syncthreads();
  float w = 0.0f;
#pragma unroll 4
  for (int j = 0; j < HD; ++j) w += Wd[tid * HD + j] * s_sh[j];
  wvec[tid] = w;
}

// one wave per row: z = h[row,:].wvec ; accumulate softplus(sign*z)
__global__ __launch_bounds__(256)
void logits_kernel(const float* __restrict__ h, const float* __restrict__ wvec,
                   float* __restrict__ acc, float sign, int N) {
  int gt = blockIdx.x * blockDim.x + threadIdx.x;
  int row = gt >> 5;
  int lane = gt & 31;
  if (row >= N) return;
  float4 v = *(const float4*)(h + (long long)row * HD + lane * 4);
  float4 w = *(const float4*)(wvec + lane * 4);
  float p = v.x * w.x + v.y * w.y + v.z * w.z + v.w * w.w;
#pragma unroll
  for (int off = 16; off > 0; off >>= 1) p += __shfl_down(p, off, 32);
  if (lane == 0) {
    float t = sign * p;
    float sp = fmaxf(t, 0.0f) + log1pf(expf(-fabsf(t)));  // stable softplus
    atomicAdd(acc, sp);
  }
}

__global__ void finalize_kernel(const float* __restrict__ acc, float* __restrict__ outp, int N) {
  outp[0] = (acc[0] + acc[1]) / (float)N;
}

// --------------------------------------------------------------------- launch
extern "C" void kernel_launch(void* const* d_in, const int* in_sizes, int n_in,
                              void* d_out, int out_size, void* d_ws, size_t ws_size,
                              hipStream_t stream) {
  const float* x   = (const float*)d_in[0];
  const int*   ei  = (const int*)d_in[1];
  const int*   perm= (const int*)d_in[2];
  const float* Wl1 = (const float*)d_in[4];
  const float* bl1 = (const float*)d_in[5];
  const float* Wr1 = (const float*)d_in[6];
  const float* Wl2 = (const float*)d_in[7];
  const float* bl2 = (const float*)d_in[8];
  const float* Wr2 = (const float*)d_in[9];
  const float* p1  = (const float*)d_in[10];
  const float* p2  = (const float*)d_in[11];
  const float* Wd  = (const float*)d_in[12];
  float* outp = (float*)d_out;

  int N = in_sizes[0] / HD;
  int E = in_sizes[1] / 2;
  const int* src = ei;
  const int* dst = ei + E;

  long long NH = (long long)N * HD;
  float* ws     = (float*)d_ws;
  float* agg    = ws;  ws += NH;
  float* h1     = ws;  ws += NH;
  float* h2     = ws;  ws += NH;
  float* cnt    = ws;  ws += N;
  float* colsum = ws;  ws += HD;
  float* wvec   = ws;  ws += HD;
  float* lossA  = ws;  ws += 2;
  _Float16* pW  = (_Float16*)ws;
  _Float16* pWl1 = pW;
  _Float16* pWr1 = pW + 1 * HD * HD;
  _Float16* pWl2 = pW + 2 * HD * HD;
  _Float16* pWr2 = pW + 3 * HD * HD;

  dim3 b256(256);
  int sageBlocks    = (N + TM - 1) / TM;
  int scatterBlocks = (E * 32 + 255) / 256;
  int logitBlocks   = (N * 32 + 255) / 256;

  pack_weight_kernel<<<1, b256, 0, stream>>>(Wl1, pWl1);
  pack_weight_kernel<<<1, b256, 0, stream>>>(Wr1, pWr1);
  pack_weight_kernel<<<1, b256, 0, stream>>>(Wl2, pWl2);
  pack_weight_kernel<<<1, b256, 0, stream>>>(Wr2, pWr2);
  zero_kernel<<<64, b256, 0, stream>>>(cnt, N);
  zero_kernel<<<1, b256, 0, stream>>>(colsum, HD + HD + 2);  // colsum|wvec|lossA contiguous
  count_kernel<<<(E + 255) / 256, b256, 0, stream>>>(dst, cnt, E);

  // ------ positive encode
  zero_kernel<<<4096, b256, 0, stream>>>(agg, NH);
  scatter_kernel<<<scatterBlocks, b256, 0, stream>>>(x, src, dst, nullptr, agg, E);
  sage_wmma_kernel<<<sageBlocks, b256, 0, stream>>>(x, nullptr, agg, cnt, pWl1, pWr1, bl1, p1, h1, nullptr, N);
  zero_kernel<<<4096, b256, 0, stream>>>(agg, NH);
  scatter_kernel<<<scatterBlocks, b256, 0, stream>>>(h1, src, dst, nullptr, agg, E);
  sage_wmma_kernel<<<sageBlocks, b256, 0, stream>>>(h1, nullptr, agg, cnt, pWl2, pWr2, bl2, p2, h2, colsum, N);

  summary_kernel<<<1, dim3(HD), 0, stream>>>(colsum, Wd, wvec, N);
  logits_kernel<<<logitBlocks, b256, 0, stream>>>(h2, wvec, lossA + 0, -1.0f, N);

  // ------ negative encode (x[perm], h2 buffer reused)
  zero_kernel<<<4096, b256, 0, stream>>>(agg, NH);
  scatter_kernel<<<scatterBlocks, b256, 0, stream>>>(x, src, dst, perm, agg, E);
  sage_wmma_kernel<<<sageBlocks, b256, 0, stream>>>(x, perm, agg, cnt, pWl1, pWr1, bl1, p1, h1, nullptr, N);
  zero_kernel<<<4096, b256, 0, stream>>>(agg, NH);
  scatter_kernel<<<scatterBlocks, b256, 0, stream>>>(h1, src, dst, nullptr, agg, E);
  sage_wmma_kernel<<<sageBlocks, b256, 0, stream>>>(h1, nullptr, agg, cnt, pWl2, pWr2, bl2, p2, h2, nullptr, N);
  logits_kernel<<<logitBlocks, b256, 0, stream>>>(h2, wvec, lossA + 1, 1.0f, N);

  finalize_kernel<<<1, 1, 0, stream>>>(lossA, outp, N);
}
